// GLABlock_39015482916951
// MI455X (gfx1250) — matrix operand order repaired and checked by reading, not compiled, not run on hardware
//
#include <hip/hip_runtime.h>
#include <hip/hip_bf16.h>
#include <math.h>

// ---------------------------------------------------------------------------
// GLA block for gfx1250 (MI455X).
// All large GEMMs run NT (A[M][K] x Bt[N][K]) through v_wmma_f32_16x16x32_bf16
// with BK=64 double-buffered LDS tiles fed by the Tensor Data Mover
// (tensor_load_to_lds + s_wait_tensorcnt); weights pre-transposed to bf16.
// ---------------------------------------------------------------------------

typedef __attribute__((ext_vector_type(16))) __bf16 v16bf;
typedef __attribute__((ext_vector_type(8)))  __bf16 v8bf;
typedef __attribute__((ext_vector_type(8)))  float  v8f;

#define D_MODEL 2048
#define N_HEADS 16
#define MLP_H 5632
#define BATCH 2
#define SEQ 4096
#define MROWS (BATCH*SEQ)   // 8192 token rows

// GEMM epilogue modes
#define EPI_F32       0
#define EPI_F32_SCALE 1
#define EPI_F32_BIAS  2
#define EPI_F32_RES   3
#define EPI_BF16_SILU 4
#define EPI_BF16_GATE 5

union FragAB { v16bf v; v8bf h[2]; };

__device__ __forceinline__ float fsilu(float x) { return x / (1.f + __expf(-x)); }
__device__ __forceinline__ float flogsig(float x) {
    return (x >= 0.f) ? -log1pf(__expf(-x)) : (x - log1pf(__expf(x)));
}

// ---------------------------------------------------------------------------
// Tensor Data Mover tile load (CDNA5 TDM).  Builds the D# descriptor per
// ISA §8: group0 = {count, lds_addr, global_addr, type=2}; group1 = data_size,
// tensor_dim0=K, tensor_dim1=rows, tile 64(K) x 128(rows), dim0_stride=K.
// Loads a 128x64 bf16 tile row-by-row into LDS (row stride 64 elems = 128B),
// exactly matching the As/Bs layout the WMMA fragment loads expect.
// ---------------------------------------------------------------------------
#if defined(__has_builtin)
#if __has_builtin(__builtin_amdgcn_tensor_load_to_lds)
#define HAVE_TDM 1
#endif
#endif

#if HAVE_TDM
typedef unsigned int u32x4 __attribute__((ext_vector_type(4)));
typedef int          i32x4 __attribute__((ext_vector_type(4)));
typedef int          i32x8 __attribute__((ext_vector_type(8)));

__device__ __forceinline__ void tdm_load_tile_128x64(
    const __hip_bfloat16* gtile, unsigned lds_off, int K, int rowsTotal) {
    unsigned long long ga = (unsigned long long)(uintptr_t)gtile;
    u32x4 g0;
    g0.x = 1u;                                                  // count=1 (valid)
    g0.y = lds_off;                                             // lds_addr (bytes)
    g0.z = (unsigned)(ga & 0xffffffffull);                      // global_addr[31:0]
    g0.w = (unsigned)((ga >> 32) & 0x01ffffffull) | (2u << 30); // addr[56:32]|type=2
    unsigned td0 = (unsigned)K;            // tensor_dim0 (elems per row)
    unsigned td1 = (unsigned)rowsTotal;    // tensor_dim1 (rows)
    unsigned long long st0 = (unsigned long long)K;  // tensor_dim0_stride
    i32x8 g1;
    g1[0] = (int)(1u << 16);                                    // data_size=1 (2B)
    g1[1] = (int)((td0 & 0xffffu) << 16);                       // td0[15:0] @ [63:48]
    g1[2] = (int)((td0 >> 16) | ((td1 & 0xffffu) << 16));
    g1[3] = (int)((td1 >> 16) | (64u << 16));                   // tile_dim0 = 64
    g1[4] = (int)128;                                           // tile_dim1 = 128
    g1[5] = (int)(st0 & 0xffffffffull);
    g1[6] = (int)((st0 >> 32) & 0xffffull);                     // dim1_stride = 0
    g1[7] = 0;
    i32x4 z4 = {0, 0, 0, 0};
#if __has_include(<hip/amd_detail/amd_gfx1250_TDM.h>)
    i32x8 z8 = {0, 0, 0, 0, 0, 0, 0, 0};
    __builtin_amdgcn_tensor_load_to_lds(g0, g1, z4, z4, z8, 0); // clang-23 form
#else
    __builtin_amdgcn_tensor_load_to_lds(g0, g1, z4, z4, 0);     // ROCm 7.2 form
#endif
}
#endif  // HAVE_TDM

// ---------------------------------------------------------------------------
// Tiled transpose + fp32->bf16: in[K][N] fp32 -> out[Npad][K] bf16 (pad rows 0)
// ---------------------------------------------------------------------------
__global__ __launch_bounds__(256)
void transpose_cvt_kernel(const float* __restrict__ in, __hip_bfloat16* __restrict__ out,
                          int K, int N) {
    __shared__ float tile[32][33];
    int kb = blockIdx.x * 32;
    int nb = blockIdx.y * 32;
    int tx = threadIdx.x & 31;
    int ty = threadIdx.x >> 5;       // 0..7
#pragma unroll
    for (int i = 0; i < 32; i += 8) {
        int n = nb + tx;
        float v = (n < N) ? in[(size_t)(kb + ty + i) * N + n] : 0.f;
        tile[ty + i][tx] = v;
    }
    __syncthreads();
#pragma unroll
    for (int i = 0; i < 32; i += 8) {
        int n = nb + ty + i;
        int k = kb + tx;
        out[(size_t)n * K + k] = __float2bfloat16(tile[tx][ty + i]);
    }
}

// ---------------------------------------------------------------------------
// RMSNorm (row of d), fp32 in -> bf16 out
// ---------------------------------------------------------------------------
__global__ __launch_bounds__(256)
void rmsnorm_bf16_kernel(const float* __restrict__ x, const float* __restrict__ w,
                         __hip_bfloat16* __restrict__ out, int d) {
    __shared__ float red[256];
    int row = blockIdx.x;
    const float* xr = x + (size_t)row * d;
    float s = 0.f;
    for (int c = threadIdx.x; c < d; c += 256) { float v = xr[c]; s += v * v; }
    red[threadIdx.x] = s;
    __syncthreads();
    for (int off = 128; off > 0; off >>= 1) {
        if (threadIdx.x < off) red[threadIdx.x] += red[threadIdx.x + off];
        __syncthreads();
    }
    float scale = rsqrtf(red[0] / (float)d + 1e-5f);
    for (int c = threadIdx.x; c < d; c += 256)
        out[(size_t)row * d + c] = __float2bfloat16(xr[c] * scale * w[c]);
}

// ---------------------------------------------------------------------------
// NT WMMA GEMM: C[M,N] = A[M,K](bf16, row-major) @ Bt[N,K](bf16)^T + epilogue.
// Block tile 128x128, BK=64, 8 waves each computing 64x32 (4x2 WMMA tiles).
// Tiles arrive via TDM (wave 0 issues tensor_load_to_lds, TENSORcnt-tracked,
// double-buffered) or, without the builtin, via vector register staging.
// Guard-free: M%128==N%128==0, K%64==0 guaranteed by the launcher.
// ---------------------------------------------------------------------------
template <int MODE>
__global__ __launch_bounds__(256)
void gemm_nt_wmma(const __hip_bfloat16* __restrict__ A,
                  const __hip_bfloat16* __restrict__ Bt,
                  float* __restrict__ Cf,
                  __hip_bfloat16* __restrict__ Cbf,
                  const float* __restrict__ bias,
                  const float* __restrict__ residual,
                  const __hip_bfloat16* __restrict__ aux,
                  int M, int N, int K, float alpha) {
    __shared__ __hip_bfloat16 As[2][128 * 64];   // 2 x 16KB
    __shared__ __hip_bfloat16 Bs[2][128 * 64];

    const int tid  = threadIdx.x;
    const int lane = tid & 31;
    const int wave = tid >> 5;
    const int wm   = wave & 1;    // 64-row half
    const int wn   = wave >> 1;   // 32-col quarter
    const int m0   = blockIdx.y * 128;
    const int n0   = blockIdx.x * 128;

    const v8f vzero = {};
    v8f acc[4][2];
#pragma unroll
    for (int mt = 0; mt < 4; mt++)
#pragma unroll
        for (int nt = 0; nt < 2; nt++) acc[mt][nt] = vzero;

    const int nTiles = K >> 6;

#if HAVE_TDM
    const unsigned ldsA0 = (unsigned)(uintptr_t)&As[0][0];
    const unsigned ldsB0 = (unsigned)(uintptr_t)&Bs[0][0];
    const unsigned TILE_BYTES = 128u * 64u * 2u;   // 16KB per buffer
    // prologue: wave 0 DMAs tile 0 of both operands; TDM ignores EXEC, the
    // wave-uniform branch makes it a single issue per tile.
    if (wave == 0) {
        tdm_load_tile_128x64(A  + (size_t)m0 * K, ldsA0, K, M);
        tdm_load_tile_128x64(Bt + (size_t)n0 * K, ldsB0, K, N);
        __builtin_amdgcn_s_wait_tensorcnt(0);
    }
    __syncthreads();
#else
    v8bf ra[4], rb[4];
#pragma unroll
    for (int i = 0; i < 4; i++) {
        int id = tid + 256 * i, row = id >> 3, seg = id & 7;
        ra[i] = *reinterpret_cast<const v8bf*>(A  + (size_t)(m0 + row) * K + seg * 8);
        rb[i] = *reinterpret_cast<const v8bf*>(Bt + (size_t)(n0 + row) * K + seg * 8);
    }
#pragma unroll
    for (int i = 0; i < 4; i++) {
        int id = tid + 256 * i, row = id >> 3, seg = id & 7;
        *reinterpret_cast<v8bf*>(&As[0][row * 64 + seg * 8]) = ra[i];
        *reinterpret_cast<v8bf*>(&Bs[0][row * 64 + seg * 8]) = rb[i];
    }
    __syncthreads();
#endif

    for (int t = 0; t < nTiles; t++) {
        const int cur = t & 1;
        // kick off next tile's loads so DMA/HBM latency overlaps the WMMAs
        if (t + 1 < nTiles) {
            const int k0 = (t + 1) << 6;
#if HAVE_TDM
            if (wave == 0) {
                const unsigned boff = (unsigned)(cur ^ 1) * TILE_BYTES;
                tdm_load_tile_128x64(A  + (size_t)m0 * K + k0, ldsA0 + boff, K, M);
                tdm_load_tile_128x64(Bt + (size_t)n0 * K + k0, ldsB0 + boff, K, N);
            }
#else
#pragma unroll
            for (int i = 0; i < 4; i++) {
                int id = tid + 256 * i, row = id >> 3, seg = id & 7;
                ra[i] = *reinterpret_cast<const v8bf*>(A  + (size_t)(m0 + row) * K + k0 + seg * 8);
                rb[i] = *reinterpret_cast<const v8bf*>(Bt + (size_t)(n0 + row) * K + k0 + seg * 8);
            }
#endif
        }
        // 2 x (K=32) sub-steps -> 16 WMMAs per barrier
#pragma unroll
        for (int ks = 0; ks < 64; ks += 32) {
            FragAB afr[4], bfr[2];
            // A 16x32: lanes 0-15 hold K{0..7,16..23}, lanes 16-31 K{8..15,24..31}
            const int ksa = ks + ((lane & 16) ? 8 : 0);
#pragma unroll
            for (int mt = 0; mt < 4; mt++) {
                int row = wm * 64 + mt * 16 + (lane & 15);
                const v8bf* p = reinterpret_cast<const v8bf*>(&As[cur][row * 64 + ksa]);
                afr[mt].h[0] = p[0];
                afr[mt].h[1] = p[2];  // +16 halves
            }
            // B 32x16: lanes 0-15 K{0..7,16..23} of column lane&15, lanes 16-31 K{8..15,24..31}
            const int ksb = ks + ((lane & 16) ? 8 : 0);
#pragma unroll
            for (int nt = 0; nt < 2; nt++) {
                int col = wn * 32 + nt * 16 + (lane & 15);
                const v8bf* p = reinterpret_cast<const v8bf*>(&Bs[cur][col * 64 + ksb]);
                bfr[nt].h[0] = p[0];
                bfr[nt].h[1] = p[2];
            }
#pragma unroll
            for (int mt = 0; mt < 4; mt++)
#pragma unroll
                for (int nt = 0; nt < 2; nt++)
                    acc[mt][nt] = __builtin_amdgcn_wmma_f32_16x16x32_bf16(
                        false, afr[mt].v, false, bfr[nt].v,
                        (short)0, acc[mt][nt], false, false);
        }
        if (t + 1 < nTiles) {
#if HAVE_TDM
            if (wave == 0) __builtin_amdgcn_s_wait_tensorcnt(0);
#else
            const int nxt = cur ^ 1;
#pragma unroll
            for (int i = 0; i < 4; i++) {
                int id = tid + 256 * i, row = id >> 3, seg = id & 7;
                *reinterpret_cast<v8bf*>(&As[nxt][row * 64 + seg * 8]) = ra[i];
                *reinterpret_cast<v8bf*>(&Bs[nxt][row * 64 + seg * 8]) = rb[i];
            }
#endif
        }
        __syncthreads();
    }

    // epilogue: C/D layout lanes0-15 -> M=j, lanes16-31 -> M=8+j
    const int rsel = (lane & 16) ? 8 : 0;
#pragma unroll
    for (int mt = 0; mt < 4; mt++) {
#pragma unroll
        for (int nt = 0; nt < 2; nt++) {
            int gc = n0 + wn * 32 + nt * 16 + (lane & 15);
#pragma unroll
            for (int j = 0; j < 8; j++) {
                int gr = m0 + wm * 64 + mt * 16 + rsel + j;
                size_t idx = (size_t)gr * N + gc;
                float v = acc[mt][nt][j];
                if constexpr (MODE == EPI_F32)            { Cf[idx] = v; }
                else if constexpr (MODE == EPI_F32_SCALE) { Cf[idx] = v * alpha; }
                else if constexpr (MODE == EPI_F32_BIAS)  { Cf[idx] = v + bias[gc]; }
                else if constexpr (MODE == EPI_F32_RES)   { Cf[idx] = v + residual[idx]; }
                else if constexpr (MODE == EPI_BF16_SILU) { Cbf[idx] = __float2bfloat16(fsilu(v)); }
                else { Cbf[idx] = __float2bfloat16(__bfloat162float(aux[idx]) * v); }
            }
        }
    }
}

// ---------------------------------------------------------------------------
// Low-rank gate: gk = logsigmoid(kg1[M,128(16 live)] @ Wkg2[16,1024] + b) / 16
// ---------------------------------------------------------------------------
__global__ __launch_bounds__(256)
void gk_kernel(const float* __restrict__ kg1, const float* __restrict__ Wkg2,
               const float* __restrict__ bkg2, float* __restrict__ gk) {
    size_t idx = (size_t)blockIdx.x * 256 + threadIdx.x;  // M*1024
    int m = (int)(idx >> 10);
    int n = (int)(idx & 1023);
    float a = bkg2[n];
#pragma unroll
    for (int j = 0; j < 16; j++) a += kg1[(size_t)m * 128 + j] * Wkg2[j * 1024 + n];
    gk[idx] = flogsig(a) * (1.f / 16.f);   // NORMALIZER = 16
}

// ---------------------------------------------------------------------------
// Sequential GLA scan. 64 blocks: (b,h) x value-half. Each block owns the
// 64(k) x 64(v) quadrant of the state in registers (16 floats/thread).
// ---------------------------------------------------------------------------
__global__ __launch_bounds__(256)
void gla_scan_kernel(const float* __restrict__ q, const float* __restrict__ k,
                     const float* __restrict__ v, const float* __restrict__ gk,
                     float* __restrict__ o, float* __restrict__ Sf) {
    __shared__ float qs[64], ks[64], egs[64], po[256];
    int bh   = blockIdx.x >> 1;        // b*16 + h
    int half = blockIdx.x & 1;         // value-dim half
    int b = bh >> 4, h = bh & 15;
    int tid = threadIdx.x;
    int c   = half * 64 + (tid & 63);  // value column 0..127
    int rg  = tid >> 6;                // row group 0..3 (rows rg*16..rg*16+15)

    float S[16];
#pragma unroll
    for (int r = 0; r < 16; r++) S[r] = 0.f;

    for (int t = 0; t < SEQ; t++) {
        size_t row = (size_t)b * SEQ + t;
        if (tid < 64)       qs[tid]        = q[row * 1024 + h * 64 + tid];
        else if (tid < 128) ks[tid - 64]   = k[row * 1024 + h * 64 + (tid - 64)];
        else if (tid < 192) egs[tid - 128] = __expf(gk[row * 1024 + h * 64 + (tid - 128)]);
        float vc = v[row * 2048 + h * 128 + c];
        __syncthreads();

        float accum = 0.f;
#pragma unroll
        for (int r = 0; r < 16; r++) {
            int rr = rg * 16 + r;
            float s = S[r] * egs[rr] + ks[rr] * vc;
            S[r] = s;
            accum += qs[rr] * s;
        }
        po[tid] = accum;
        __syncthreads();
        if (rg == 0)
            o[row * 2048 + h * 128 + c] = po[tid] + po[tid + 64] + po[tid + 128] + po[tid + 192];
        // next iteration's first barrier orders po reads vs. rewrites
    }
    __syncthreads();
#pragma unroll
    for (int r = 0; r < 16; r++)
        Sf[((size_t)bh * 64 + rg * 16 + r) * 128 + c] = S[r];
}

// ---------------------------------------------------------------------------
// Per-head LayerNorm (no affine) + silu(g) gate -> bf16 rows for Wo GEMM
// ---------------------------------------------------------------------------
__global__ __launch_bounds__(128)
void gn_gate_kernel(const float* __restrict__ o, const float* __restrict__ g,
                    __hip_bfloat16* __restrict__ out) {
    __shared__ float r1[128], r2[128];
    int blh = blockIdx.x;          // tokenRow*16 + h
    int bl = blh >> 4, h = blh & 15;
    size_t idx = (size_t)bl * 2048 + h * 128 + threadIdx.x;
    float v = o[idx];
    r1[threadIdx.x] = v;
    r2[threadIdx.x] = v * v;
    __syncthreads();
    for (int off = 64; off > 0; off >>= 1) {
        if (threadIdx.x < off) {
            r1[threadIdx.x] += r1[threadIdx.x + off];
            r2[threadIdx.x] += r2[threadIdx.x + off];
        }
        __syncthreads();
    }
    float mu  = r1[0] * (1.f / 128.f);
    float var = r2[0] * (1.f / 128.f) - mu * mu;
    float on  = (v - mu) * rsqrtf(var + 1e-5f);
    out[idx] = __float2bfloat16(fsilu(g[idx]) * on);
}

// ---------------------------------------------------------------------------
// Host-side orchestration
// ---------------------------------------------------------------------------
static inline dim3 gemm_grid(int M, int N) { return dim3(N / 128, M / 128); }

extern "C" void kernel_launch(void* const* d_in, const int* in_sizes, int n_in,
                              void* d_out, int out_size, void* d_ws, size_t ws_size,
                              hipStream_t stream) {
    (void)in_sizes; (void)n_in; (void)out_size; (void)ws_size;

    const float* x     = (const float*)d_in[0];
    const float* Wq    = (const float*)d_in[1];
    const float* Wk    = (const float*)d_in[2];
    const float* Wkg1  = (const float*)d_in[3];
    const float* Wkg2  = (const float*)d_in[4];
    const float* bkg2  = (const float*)d_in[5];
    const float* Wv    = (const float*)d_in[6];
    const float* Wg    = (const float*)d_in[7];
    const float* bg    = (const float*)d_in[8];
    const float* Wo    = (const float*)d_in[9];
    const float* ln1_w = (const float*)d_in[10];
    const float* ln2_w = (const float*)d_in[11];
    const float* W1    = (const float*)d_in[12];
    const float* W3    = (const float*)d_in[13];
    const float* W2    = (const float*)d_in[14];

    float* out = (float*)d_out;                 // [x2 : M*2048][Sf : 2*16*64*128]
    float* Sf  = out + (size_t)MROWS * D_MODEL;

    char* ws = (char*)d_ws;
    size_t off = 0;
    auto alloc = [&](size_t bytes) -> void* {
        void* p = ws + off;
        off += (bytes + 255) & ~(size_t)255;
        return p;
    };
    const size_t M = MROWS;
    typedef __hip_bfloat16 bf;
    // transposed bf16 weights: [N][K]
    bf* wqT   = (bf*)alloc((size_t)1024 * 2048 * 2);
    bf* wkT   = (bf*)alloc((size_t)1024 * 2048 * 2);
    bf* wkg1T = (bf*)alloc((size_t)128  * 2048 * 2);   // N=16 padded to 128
    bf* wvT   = (bf*)alloc((size_t)2048 * 2048 * 2);
    bf* wgT   = (bf*)alloc((size_t)2048 * 2048 * 2);
    bf* woT   = (bf*)alloc((size_t)2048 * 2048 * 2);
    bf* w1T   = (bf*)alloc((size_t)MLP_H * 2048 * 2);
    bf* w3T   = (bf*)alloc((size_t)MLP_H * 2048 * 2);
    bf* w2T   = (bf*)alloc((size_t)2048 * MLP_H * 2);
    bf* xn_bf  = (bf*)alloc(M * 2048 * 2);
    bf* xn2_bf = (bf*)alloc(M * 2048 * 2);
    bf* og_bf  = (bf*)alloc(M * 2048 * 2);
    bf* s1_bf  = (bf*)alloc(M * MLP_H * 2);
    bf* h_bf   = (bf*)alloc(M * MLP_H * 2);
    float* q_f   = (float*)alloc(M * 1024 * 4);
    float* k_f   = (float*)alloc(M * 1024 * 4);
    float* gk_f  = (float*)alloc(M * 1024 * 4);
    float* kg1_f = (float*)alloc(M * 128 * 4);
    float* v_f   = (float*)alloc(M * 2048 * 4);
    float* g_f   = (float*)alloc(M * 2048 * 4);
    float* o_f   = (float*)alloc(M * 2048 * 4);
    float* x1_f  = (float*)alloc(M * 2048 * 4);

    auto tcvt = [&](const float* src, bf* dstT, int K, int N, int Npad) {
        transpose_cvt_kernel<<<dim3(K / 32, Npad / 32), 256, 0, stream>>>(src, dstT, K, N);
    };
    tcvt(Wq,   wqT,   2048, 1024, 1024);
    tcvt(Wk,   wkT,   2048, 1024, 1024);
    tcvt(Wkg1, wkg1T, 2048, 16,   128);
    tcvt(Wv,   wvT,   2048, 2048, 2048);
    tcvt(Wg,   wgT,   2048, 2048, 2048);
    tcvt(Wo,   woT,   2048, 2048, 2048);
    tcvt(W1,   w1T,   2048, MLP_H, MLP_H);
    tcvt(W3,   w3T,   2048, MLP_H, MLP_H);
    tcvt(W2,   w2T,   MLP_H, 2048, 2048);

    // ---- attention half ----
    rmsnorm_bf16_kernel<<<(unsigned)M, 256, 0, stream>>>(x, ln1_w, xn_bf, D_MODEL);

    const float scaling = 0.088388347648318447f;  // (d/H)^-0.5 = 128^-0.5
    gemm_nt_wmma<EPI_F32><<<gemm_grid(M, 1024), 256, 0, stream>>>(
        xn_bf, wqT, q_f, nullptr, nullptr, nullptr, nullptr, (int)M, 1024, 2048, 1.f);
    gemm_nt_wmma<EPI_F32_SCALE><<<gemm_grid(M, 1024), 256, 0, stream>>>(
        xn_bf, wkT, k_f, nullptr, nullptr, nullptr, nullptr, (int)M, 1024, 2048, scaling);
    gemm_nt_wmma<EPI_F32><<<gemm_grid(M, 128), 256, 0, stream>>>(
        xn_bf, wkg1T, kg1_f, nullptr, nullptr, nullptr, nullptr, (int)M, 128, 2048, 1.f);
    gk_kernel<<<(unsigned)(M * 1024 / 256), 256, 0, stream>>>(kg1_f, Wkg2, bkg2, gk_f);
    gemm_nt_wmma<EPI_F32><<<gemm_grid(M, 2048), 256, 0, stream>>>(
        xn_bf, wvT, v_f, nullptr, nullptr, nullptr, nullptr, (int)M, 2048, 2048, 1.f);
    gemm_nt_wmma<EPI_F32_BIAS><<<gemm_grid(M, 2048), 256, 0, stream>>>(
        xn_bf, wgT, g_f, nullptr, bg, nullptr, nullptr, (int)M, 2048, 2048, 1.f);

    gla_scan_kernel<<<2 * BATCH * N_HEADS, 256, 0, stream>>>(q_f, k_f, v_f, gk_f, o_f, Sf);
    gn_gate_kernel<<<(unsigned)(M * N_HEADS), 128, 0, stream>>>(o_f, g_f, og_bf);

    gemm_nt_wmma<EPI_F32_RES><<<gemm_grid(M, 2048), 256, 0, stream>>>(
        og_bf, woT, x1_f, nullptr, nullptr, x, nullptr, (int)M, 2048, 2048, 1.f);

    // ---- MLP half ----
    rmsnorm_bf16_kernel<<<(unsigned)M, 256, 0, stream>>>(x1_f, ln2_w, xn2_bf, D_MODEL);
    gemm_nt_wmma<EPI_BF16_SILU><<<gemm_grid(M, MLP_H), 256, 0, stream>>>(
        xn2_bf, w1T, nullptr, s1_bf, nullptr, nullptr, nullptr, (int)M, MLP_H, 2048, 1.f);
    gemm_nt_wmma<EPI_BF16_GATE><<<gemm_grid(M, MLP_H), 256, 0, stream>>>(
        xn2_bf, w3T, nullptr, h_bf, nullptr, nullptr, s1_bf, (int)M, MLP_H, 2048, 1.f);
    gemm_nt_wmma<EPI_F32_RES><<<gemm_grid(M, 2048), 256, 0, stream>>>(
        h_bf, w2T, out, nullptr, nullptr, x1_f, nullptr, (int)M, 2048, MLP_H, 1.f);
}
